// FFT_TREND_76424648065379
// MI455X (gfx1250) — compile-verified, matrix operand order checked by reading, not run on hardware
//
#include <hip/hip_runtime.h>
#include <hip/hip_bf16.h>

typedef _Float16 half_t;
typedef __attribute__((ext_vector_type(16))) _Float16 v16h;
typedef __attribute__((ext_vector_type(8)))  _Float16 v8h;
typedef __attribute__((ext_vector_type(8)))  float    v8f;

#define BB   32
#define NN   256
#define TT   2048
#define KK   5
#define NF   1025     // rfft bins
#define MPAD 1040     // 65 * 16

// ---------------------------------------------------------------------------
// Kernel 1: build f16 DFT weight tables (cos, -sin), zero the mag accumulator.
// ---------------------------------------------------------------------------
__global__ __launch_bounds__(256) void prep_kernel(half_t* __restrict__ Wc,
                                                   half_t* __restrict__ Ws,
                                                   float*  __restrict__ mag) {
    int i = blockIdx.x * 256 + threadIdx.x;
    if (i < MPAD * TT) {
        int f = i >> 11;          // row (freq)
        int t = i & (TT - 1);     // col (time)
        half_t c = (half_t)0.f, s = (half_t)0.f;
        if (f < NF) {
            int ph = (f * t) & (TT - 1);                 // (f*t) mod T, exact
            float ang = (float)ph * (6.28318530717958647692f / (float)TT);
            float sf, cf;
            __sincosf(ang, &sf, &cf);
            c = (half_t)cf;
            s = (half_t)(-sf);
        }
        Wc[i] = c;
        Ws[i] = s;
    }
    if (i < BB * MPAD) mag[i] = 0.f;
}

// ---------------------------------------------------------------------------
// Kernel 2: mag[b,f] = sum_n |DFT(x[b,n,:])[f]| via WMMA f16 GEMMs.
// A (cos/sin) tiles are streamed into LDS with double-buffered
// GLOBAL_LOAD_ASYNC_TO_LDS_B128 (ASYNCcnt), K chunked by 128.
// Grid: (MPAD/16, B). Block: 256 threads = 8 waves; wave w owns N-cols
// [32w, 32w+32) as two 16-wide tiles with Re/Im f32 accumulators.
// ---------------------------------------------------------------------------
__global__ __launch_bounds__(256) void dft_mag_kernel(const float* __restrict__ x,
                                                      const half_t* __restrict__ Wc,
                                                      const half_t* __restrict__ Ws,
                                                      float* __restrict__ mag) {
    // [buffer][cos=0/sin=1][row*128 + t']  -> 16 KB total
    __shared__ __align__(16) half_t lA[2][2][16 * 128];

    const int mtile = blockIdx.x;          // 16-freq tile
    const int b     = blockIdx.y;          // batch
    const int tid   = threadIdx.x;
    const int lane  = tid & 31;
    const int wave  = tid >> 5;            // 0..7
    const int lm    = lane & 15;           // row/col within 16-tile
    const int h8    = (lane >> 4) << 3;    // K sub-offset for upper half-wave

    // Async-copy indexing: each thread moves one 16-byte chunk per table/chunk.
    const int arow = tid >> 4;             // 0..15 (freq row within tile)
    const int acol = (tid & 15) << 3;      // half index 0..120 step 8
    const half_t* gc = Wc + (size_t)(mtile * 16 + arow) * TT + acol;
    const half_t* gs = Ws + (size_t)(mtile * 16 + arow) * TT + acol;
    const int lbase  = arow * 128 + acol;

    auto issue = [&](int buf, int k0) {
        unsigned dc = (unsigned)(uintptr_t)&lA[buf][0][lbase];
        unsigned ds = (unsigned)(uintptr_t)&lA[buf][1][lbase];
        unsigned long long pc = (unsigned long long)(uintptr_t)(gc + k0);
        unsigned long long ps = (unsigned long long)(uintptr_t)(gs + k0);
        asm volatile("global_load_async_to_lds_b128 %0, %1, off"
                     :: "v"(dc), "v"(pc) : "memory");
        asm volatile("global_load_async_to_lds_b128 %0, %1, off"
                     :: "v"(ds), "v"(ps) : "memory");
    };

    const int n0 = wave * 32 + lm;         // B-matrix columns (channels)
    const float* x0p = x + ((size_t)b * NN + n0) * TT;
    const float* x1p = x0p + (size_t)16 * TT;     // n0 + 16

    v8f re0 = {}, im0 = {}, re1 = {}, im1 = {};

    issue(0, 0);
    for (int c = 0; c < 16; ++c) {
        if (c + 1 < 16) {
            issue((c + 1) & 1, (c + 1) * 128);
            // my chunk-c pair is complete once <=2 (the chunk-c+1 pair) remain
            asm volatile("s_wait_asynccnt 0x2" ::: "memory");
        } else {
            asm volatile("s_wait_asynccnt 0x0" ::: "memory");
        }
        __syncthreads();   // chunk c visible block-wide

        const half_t* Ac = &lA[c & 1][0][0];
        const half_t* As = &lA[c & 1][1][0];

#pragma unroll
        for (int st = 0; st < 4; ++st) {
            const int kloc = st * 32;
            const int kglb = c * 128 + kloc;

            // A fragments from LDS: two contiguous 16B chunks per table.
            v8h clo = *(const v8h*)(Ac + lm * 128 + kloc + h8);
            v8h chi = *(const v8h*)(Ac + lm * 128 + kloc + 16 + h8);
            v8h slo = *(const v8h*)(As + lm * 128 + kloc + h8);
            v8h shi = *(const v8h*)(As + lm * 128 + kloc + 16 + h8);
            v16h ac, as, b0, b1;
#pragma unroll
            for (int e = 0; e < 8; ++e) {
                ac[e] = clo[e]; ac[e + 8] = chi[e];
                as[e] = slo[e]; as[e + 8] = shi[e];
            }
            // B fragments: f32 global loads + inline cvt (merges to b128+cvt_pk)
#pragma unroll
            for (int e = 0; e < 16; ++e) {
                int t = kglb + ((e & 8) << 1) + h8 + (e & 7);
                b0[e] = (half_t)x0p[t];
                b1[e] = (half_t)x1p[t];
            }
            re0 = __builtin_amdgcn_wmma_f32_16x16x32_f16(false, ac, false, b0, (short)0, re0, false, false);
            im0 = __builtin_amdgcn_wmma_f32_16x16x32_f16(false, as, false, b0, (short)0, im0, false, false);
            re1 = __builtin_amdgcn_wmma_f32_16x16x32_f16(false, ac, false, b1, (short)0, re1, false, false);
            im1 = __builtin_amdgcn_wmma_f32_16x16x32_f16(false, as, false, b1, (short)0, im1, false, false);
        }
        __syncthreads();   // all reads of buf[c&1] done before it is refilled
    }

    // |.| then reduce over the 16 N-lanes of each half-wave; C/D layout:
    // VGPR r holds M = r (lanes 0-15) and M = r+8 (lanes 16-31), N = lane&15.
#pragma unroll
    for (int r = 0; r < 8; ++r) {
        float a = sqrtf(re0[r] * re0[r] + im0[r] * im0[r])
                + sqrtf(re1[r] * re1[r] + im1[r] * im1[r]);
        a += __shfl_xor(a, 1, 16);
        a += __shfl_xor(a, 2, 16);
        a += __shfl_xor(a, 4, 16);
        a += __shfl_xor(a, 8, 16);
        if (lm == 0) {
            int fr = mtile * 16 + r + h8;
            if (fr < NF) atomicAdd(&mag[b * MPAD + fr], a);
        }
    }
}

// ---------------------------------------------------------------------------
// Kernel 3: per-batch top-5 (excluding DC), lower-index tie-break, window sizes.
// ---------------------------------------------------------------------------
__global__ __launch_bounds__(256) void topk_kernel(const float* __restrict__ mag,
                                                   int* __restrict__ ks) {
    const int b = blockIdx.x, tid = threadIdx.x;
    __shared__ float sv[MPAD];
    __shared__ float rv[256];
    __shared__ int   ri[256];
    __shared__ int   top[KK];

    for (int i = tid; i < MPAD; i += 256)
        sv[i] = (i >= 1 && i < NF) ? mag[b * MPAD + i] : -1e30f;
    __syncthreads();

    for (int r = 0; r < KK; ++r) {
        float bv = -1e30f; int bi = 0x7fffffff;
        for (int i = tid; i < MPAD; i += 256) {
            float v = sv[i];
            if (v > bv || (v == bv && i < bi)) { bv = v; bi = i; }
        }
        rv[tid] = bv; ri[tid] = bi;
        __syncthreads();
        for (int s = 128; s > 0; s >>= 1) {
            if (tid < s) {
                float v = rv[tid + s]; int i2 = ri[tid + s];
                if (v > rv[tid] || (v == rv[tid] && i2 < ri[tid])) { rv[tid] = v; ri[tid] = i2; }
            }
            __syncthreads();
        }
        if (tid == 0) { top[r] = ri[0]; sv[ri[0]] = -1e30f; }
        __syncthreads();
    }

    if (tid < KK) {
        int idx = top[tid];
        float freq = (float)idx * (120.0f / 2048.0f);
        ks[b * KK + tid] = (int)floorf(2048.0f / freq);
    }
}

// ---------------------------------------------------------------------------
// Kernel 4: multi-scale centered moving mean with edge replication.
// One block per (b, n) row: build prefix sum C[0..T] in LDS, emit 5 windows.
// ---------------------------------------------------------------------------
__global__ __launch_bounds__(256) void smooth_kernel(const float* __restrict__ x,
                                                     const int* __restrict__ ks,
                                                     float* __restrict__ out) {
    const int n = blockIdx.x, b = blockIdx.y, tid = threadIdx.x;
    __shared__ float Cs[TT + 1];
    __shared__ float part[256];

    const float* xr = x + ((size_t)b * NN + n) * TT;
    float loc[8];
    float run = 0.f;
    const int base = tid * 8;
#pragma unroll
    for (int i = 0; i < 8; ++i) { run += xr[base + i]; loc[i] = run; }
    part[tid] = run;
    __syncthreads();
    if (tid == 0) {
        float acc = 0.f;
        for (int j = 0; j < 256; ++j) { float v = part[j]; part[j] = acc; acc += v; }
    }
    __syncthreads();
    const float off = part[tid];
    if (tid == 0) Cs[0] = 0.f;
#pragma unroll
    for (int i = 0; i < 8; ++i) Cs[base + i + 1] = off + loc[i];
    __syncthreads();

    const float xfirst = Cs[1];
    const float xlast  = Cs[TT] - Cs[TT - 1];
    float* ob = out + (((size_t)b * NN + n) * KK) * TT;

    for (int j = 0; j < KK; ++j) {
        const int k   = ks[b * KK + j];
        const int p   = (k - 1) >> 1;
        const float inv = 1.f / (float)k;
        const bool even = (k & 1) == 0;
        for (int t = tid; t < TT; t += 256) {
            int te  = even ? min(t, TT - 2) : t;
            int a   = te - p;
            int bb2 = a + k - 1;
            int nf  = min(max(-a, 0), k);
            int ne  = min(max(bb2 - (TT - 1), 0), k);
            int lo  = min(max(a, 0), TT);
            int hi  = min(max(bb2 + 1, 0), TT);
            float mid = Cs[hi] - Cs[lo];
            ob[(size_t)j * TT + t] = ((float)nf * xfirst + mid + (float)ne * xlast) * inv;
        }
    }
}

// ---------------------------------------------------------------------------
extern "C" void kernel_launch(void* const* d_in, const int* in_sizes, int n_in,
                              void* d_out, int out_size, void* d_ws, size_t ws_size,
                              hipStream_t stream) {
    (void)in_sizes; (void)n_in; (void)out_size; (void)ws_size;
    const float* x = (const float*)d_in[0];
    float* out = (float*)d_out;

    // Workspace layout (~8.7 MB total)
    half_t* Wc  = (half_t*)d_ws;                       // MPAD*TT halves
    half_t* Ws  = Wc + (size_t)MPAD * TT;              // MPAD*TT halves
    float*  mag = (float*)(Ws + (size_t)MPAD * TT);    // BB*MPAD floats
    int*    ksb = (int*)(mag + (size_t)BB * MPAD);     // BB*KK ints

    const int prepN = MPAD * TT;
    hipLaunchKernelGGL(prep_kernel, dim3((prepN + 255) / 256), dim3(256), 0, stream,
                       Wc, Ws, mag);
    hipLaunchKernelGGL(dft_mag_kernel, dim3(MPAD / 16, BB), dim3(256), 0, stream,
                       x, Wc, Ws, mag);
    hipLaunchKernelGGL(topk_kernel, dim3(BB), dim3(256), 0, stream, mag, ksb);
    hipLaunchKernelGGL(smooth_kernel, dim3(NN, BB), dim3(256), 0, stream, x, ksb, out);
}
